// PhotonicDelayReservoirAdaptive_88295937671439
// MI455X (gfx1250) — matrix operand order repaired and checked by reading, not compiled, not run on hardware
//
#include <hip/hip_runtime.h>
#include <math.h>

typedef __attribute__((ext_vector_type(16))) _Float16 v16h;
typedef __attribute__((ext_vector_type(8)))  float    v8f;

#define B_DIM   64
#define T_DIM   2048
#define R_DIM   256
#define NTAPS   5
#define KSTEPS  40                       // 1280 / 32
#define NBLK    8                        // 8 workgroups, 32 output columns each
#define NCOLS   32
#define TR      ((size_t)T_DIM * R_DIM)
#define LEAK    0.1f

// dynamic LDS layout (CDNA5: up to 320KB per workgroup)
#define BLDS_BYTES  (KSTEPS * 2 * 512 * 2)        // 81920 B : B fragments, 2 n-tiles
#define RED_BYTES   (4 * 32 * 16 * 4)             // 8192 B  : K-split reduction
#define SMEM_BYTES  (BLDS_BYTES + RED_BYTES + 256)

__global__ void rsv_init_kernel(int* bar) {
    if (threadIdx.x == 0 && blockIdx.x == 0) *bar = 0;
}

__global__ void __launch_bounds__(256, 1)
rsv_main_kernel(const float* __restrict__ x,
                const float* __restrict__ W_in,
                const float* __restrict__ W_fb,
                const float* __restrict__ tap_weights,
                const float* __restrict__ bias,
                float* __restrict__ out,
                int* __restrict__ bar)
{
    extern __shared__ char smem[];
    _Float16* Blds = (_Float16*)smem;                        // [kk][ntile][512]
    float*    Red  = (float*)(smem + BLDS_BYTES);            // [wave0..3][lane][16]
    float*    xs   = (float*)(smem + BLDS_BYTES + RED_BYTES);// [64]

    const int tid  = threadIdx.x;
    const int lane = tid & 31;
    const int wave = tid >> 5;
    const int g    = wave >> 2;          // K-split group: 0 -> taps 0..2, 1 -> taps 3..4
    const int m0   = (wave & 3) * 16;    // batch tile
    const int s0   = blockIdx.x * NCOLS; // this WG's 32 output columns

    // softmax(tap_weights), folded into W_fb
    float tw[NTAPS];
    {
        float mx = -1e30f;
        for (int k = 0; k < NTAPS; ++k) mx = fmaxf(mx, tap_weights[k]);
        float sum = 0.f;
        for (int k = 0; k < NTAPS; ++k) { tw[k] = __expf(tap_weights[k] - mx); sum += tw[k]; }
        const float inv = 1.0f / sum;
        for (int k = 0; k < NTAPS; ++k) tw[k] *= inv;
    }

    // Stage scaled W_fb (both n-tiles) into LDS, exact wmma B-fragment layout:
    // index i = kk*1024 + nt*512 + l*16 + j ; K = kk*32 + (l>=16?8:0) + j + (j>=8?8:0) ; N = s0 + nt*16 + (l&15)
    for (int i = tid; i < KSTEPS * 2 * 512; i += 256) {
        const int kk  = i >> 10;
        const int nt  = (i >> 9) & 1;
        const int rem = i & 511;
        const int l   = rem >> 4;
        const int j   = rem & 15;
        const int K   = kk * 32 + ((l >> 4) << 3) + j + (j & 8);
        const int tap = K >> 8;
        const int r   = K & 255;
        const int s   = s0 + nt * 16 + (l & 15);
        Blds[i] = (_Float16)(tw[tap] * W_fb[((size_t)tap * R_DIM + r) * R_DIM + s]);
    }

    // Per-lane epilogue constants (used by wave group 0 only)
    const int   s_lane0 = s0 + (lane & 15);
    const int   s_lane1 = s_lane0 + 16;
    const float bias0 = bias[s_lane0], bias1 = bias[s_lane1];
    const float win0  = W_in[s_lane0], win1  = W_in[s_lane1];
    const int   b_lane = m0 + ((lane >> 4) << 3);

    // A-fragment addressing (all waves)
    const int aoff = (lane >> 4) << 3;   // 0 or 8
    const int arow = m0 + (lane & 15);   // batch row

    // load + f32->f16 convert of one A fragment (documented 16-bit A lane layout)
    auto loadA = [&](const float* rb, int q) -> v16h {
        const float4* p = (const float4*)(rb + (q << 5));   // 32B aligned
        const float4 q0 = p[0], q1 = p[1], q2 = p[4], q3 = p[5];
        v16h a;
        a[0]  = (_Float16)q0.x; a[1]  = (_Float16)q0.y; a[2]  = (_Float16)q0.z; a[3]  = (_Float16)q0.w;
        a[4]  = (_Float16)q1.x; a[5]  = (_Float16)q1.y; a[6]  = (_Float16)q1.z; a[7]  = (_Float16)q1.w;
        a[8]  = (_Float16)q2.x; a[9]  = (_Float16)q2.y; a[10] = (_Float16)q2.z; a[11] = (_Float16)q2.w;
        a[12] = (_Float16)q3.x; a[13] = (_Float16)q3.y; a[14] = (_Float16)q3.z; a[15] = (_Float16)q3.w;
        return a;
    };

    float h0[8], h1[8];
    #pragma unroll
    for (int v = 0; v < 8; ++v) { h0[v] = 0.f; h1[v] = 0.f; }

    __syncthreads();

    const int delays[NTAPS] = {1, 4, 24, 96, 168};
    const int tap_begin = g ? 3 : 0;
    const int tap_end   = g ? 5 : 3;

    for (int t = 0; t < T_DIM; ++t) {
        if (tid < B_DIM) xs[tid] = x[(size_t)tid * T_DIM + t];
        __syncthreads();

        v8f acc0 = {}, acc1 = {};

        for (int tap = tap_begin; tap < tap_end; ++tap) {
            const int src_t = t - delays[tap] - 1;       // states[src_t] == h[t - d]
            if (src_t < 0) continue;                      // uniform: delayed state is zero
            const float*    rowbase = out + (size_t)arow * TR + (size_t)src_t * R_DIM + aoff;
            const _Float16* bbase   = Blds + tap * 8 * 1024 + (lane << 4);

            // software-pipelined, fully unrolled: prefetch q+1's A/B fragments into
            // fresh registers before issuing q's WMMAs (double-buffered register sets;
            // removes the dscnt-0 stall between the paired WMMAs and the A-WAR nops)
            v16h a_cur  = loadA(rowbase, 0);
            v16h b0_cur = *(const v16h*)(bbase);
            v16h b1_cur = *(const v16h*)(bbase + 512);

            #pragma unroll
            for (int q = 0; q < 8; ++q) {
                v16h a_nxt = a_cur, b0_nxt = b0_cur, b1_nxt = b1_cur;
                if (q < 7) {
                    a_nxt  = loadA(rowbase, q + 1);
                    b0_nxt = *(const v16h*)(bbase + (q + 1) * 1024);
                    b1_nxt = *(const v16h*)(bbase + (q + 1) * 1024 + 512);
                }
                acc0 = __builtin_amdgcn_wmma_f32_16x16x32_f16(false, a_cur, false, b0_cur,
                                                              (short)0, acc0, false, false);
                acc1 = __builtin_amdgcn_wmma_f32_16x16x32_f16(false, a_cur, false, b1_cur,
                                                              (short)0, acc1, false, false);
                a_cur = a_nxt; b0_cur = b0_nxt; b1_cur = b1_nxt;
            }
        }

        // K-split reduction: group 1 dumps partial C tiles to LDS
        if (g == 1) {
            float* r = Red + ((wave - 4) * 32 + lane) * 16;
            *(v8f*)r       = acc0;
            *(v8f*)(r + 8) = acc1;
        }
        __syncthreads();

        if (g == 0) {
            const float* r = Red + (wave * 32 + lane) * 16;
            acc0 = acc0 + *(const v8f*)r;
            acc1 = acc1 + *(const v8f*)(r + 8);

            #pragma unroll
            for (int v = 0; v < 8; ++v) {
                const int   b  = b_lane + v;
                const float xv = xs[b];
                const float pre0 = xv * win0 + acc0[v] + bias0;
                const float pre1 = xv * win1 + acc1[v] + bias1;
                const float hn0  = (1.0f - LEAK) * h0[v] + LEAK * tanhf(pre0);
                const float hn1  = (1.0f - LEAK) * h1[v] + LEAK * tanhf(pre1);
                h0[v] = hn0; h1[v] = hn1;
                float* orow = out + (size_t)b * TR + (size_t)t * R_DIM;
                orow[s_lane0] = hn0;
                orow[s_lane1] = hn1;
            }
        }

        // Device-wide barrier across the 8 persistent blocks (release -> arrive -> spin -> acquire)
        __threadfence();
        __syncthreads();
        if (tid == 0) {
            atomicAdd(bar, 1);
            const int target = NBLK * (t + 1);
            while (__hip_atomic_load(bar, __ATOMIC_RELAXED, __HIP_MEMORY_SCOPE_AGENT) < target)
                __builtin_amdgcn_s_sleep(1);
        }
        __syncthreads();
        __threadfence();
    }
}

extern "C" void kernel_launch(void* const* d_in, const int* in_sizes, int n_in,
                              void* d_out, int out_size, void* d_ws, size_t ws_size,
                              hipStream_t stream)
{
    (void)in_sizes; (void)n_in; (void)out_size; (void)ws_size;
    const float* x   = (const float*)d_in[0];   // (64, 2048, 1)
    const float* Win = (const float*)d_in[1];   // (256, 1)
    const float* Wfb = (const float*)d_in[2];   // (5, 256, 256)
    const float* tpw = (const float*)d_in[3];   // (5,)
    const float* bia = (const float*)d_in[4];   // (256,)
    float* out = (float*)d_out;                 // (64, 2048, 256) fp32
    int*   bar = (int*)d_ws;                    // global barrier counter

    rsv_init_kernel<<<1, 1, 0, stream>>>(bar);
    rsv_main_kernel<<<NBLK, 256, SMEM_BYTES, stream>>>(x, Win, Wfb, tpw, bia, out, bar);
}
